// MeshCNNBlock_627065225595
// MI455X (gfx1250) — matrix operand order repaired and checked by reading, not compiled, not run on hardware
//
#include <hip/hip_runtime.h>
#include <math.h>

// ---------------------------------------------------------------------------
// MeshCNN conv block for gfx1250 (MI455X), wave32 + WMMA.
// y[b,o,e] = sum_{c,k} feats[b,c,k,e] * W[o,c,k]  -> BatchNorm(train) -> ReLU
// feats k: {x, n0+n2, |n0-n2|, n1+n3, |n1-n3|}, K = c*5+k in [0,80), pad to 96.
// GEMM view: W(32x80) @ feats(80 x B*E). Memory-bound => two-pass recompute,
// L2-resident gathers, LDS-staged f16 operands, v_wmma_f32_16x16x32_f16.
// Bias cancels under train-mode BN (y - mean) and is ignored.
// ---------------------------------------------------------------------------

typedef __attribute__((ext_vector_type(16))) _Float16 v16h;
typedef __attribute__((ext_vector_type(8)))  _Float16 v8h;
typedef __attribute__((ext_vector_type(8)))  float    v8f;

#define Bn      16
#define CIN     16
#define COUT    32
#define En      60000
#define KDIM    80            // CIN * 5
#define KPAD    96            // 3 k-steps of 32
#define TILE_E  128
#define NTHREADS 256
#define NWAVES  8
#define NBLKX   ((En + TILE_E - 1) / TILE_E)   // 469
#define NPART   (NBLKX * Bn)                   // 7504 partial rows
#define EPSV    1e-5f

// ---------------------------------------------------------------------------
// Phase A: stage weights (f16, K-padded) + gather indices + feats tile in LDS.
// ---------------------------------------------------------------------------
__device__ __forceinline__ void build_tile(
    const float* __restrict__ x, const int* __restrict__ gemm,
    const float* __restrict__ wgt,
    _Float16* s_w, _Float16* s_f, int* s_g,
    int b, int e0, int tid)
{
  // weight -> LDS as f16, [COUT][KPAD], zeros for K in [80,96)
  for (int i = tid; i < COUT * KPAD; i += NTHREADS) {
    int o = i / KPAD, K = i - o * KPAD;
    float wv = (K < KDIM) ? wgt[o * KDIM + K] : 0.0f;
    s_w[i] = (_Float16)wv;
  }
  // neighbor indices -> LDS
  for (int i = tid; i < TILE_E * 4; i += NTHREADS) {
    int el = i >> 2;
    int e  = e0 + el;
    s_g[i] = (e < En) ? gemm[((size_t)b * En + e) * 4 + (i & 3)] : 0;
  }
  __syncthreads();

  // feats tile: [TILE_E][KPAD] f16.  For fixed c, consecutive tid -> coalesced
  // x[b][c][e] stream; neighbor gathers hit L2 (per-batch slab = 3.84 MB).
  const float* xb = x + (size_t)b * CIN * En;
  for (int t = tid; t < TILE_E * CIN; t += NTHREADS) {
    int el = t & (TILE_E - 1);
    int c  = t >> 7;                       // t / TILE_E
    int e  = e0 + el;
    _Float16* f = &s_f[el * KPAD + c * 5];
    if (e < En) {
      const float* xc = xb + (size_t)c * En;
      float xv = xc[e];
      float a0 = xc[s_g[el * 4 + 0]];
      float a1 = xc[s_g[el * 4 + 1]];
      float a2 = xc[s_g[el * 4 + 2]];
      float a3 = xc[s_g[el * 4 + 3]];
      f[0] = (_Float16)xv;
      f[1] = (_Float16)(a0 + a2);
      f[2] = (_Float16)fabsf(a0 - a2);
      f[3] = (_Float16)(a1 + a3);
      f[4] = (_Float16)fabsf(a1 - a3);
    } else {
      // zeros: tail edges contribute 0 to sums (counts use exact B*E)
      f[0] = f[1] = f[2] = f[3] = f[4] = (_Float16)0.0f;
    }
  }
  // zero K pad region [80,96)
  for (int t = tid; t < TILE_E * (KPAD - KDIM); t += NTHREADS) {
    int el = t / (KPAD - KDIM);
    int K  = KDIM + (t % (KPAD - KDIM));
    s_f[el * KPAD + K] = (_Float16)0.0f;
  }
  __syncthreads();
}

// ---------------------------------------------------------------------------
// Phase B (per wave): one 16-edge group, 2 M-tiles (Cout 0-15 / 16-31),
// 3 K-steps of 32 -> 6 x v_wmma_f32_16x16x32_f16, f32 accumulation.
// Operand layouts used here (wave32; contiguous k order, same for A and B):
//   A 16x32 f16: lane l (M = l&15) holds K = kk + 16*(l>>4) + j
//     -> two contiguous 16-byte LDS chunks of s_w.
//   B 32x16 f16: lane l (N = l&15) holds K = kk + 16*(l>>4) + j
//     -> one contiguous 32-byte LDS read of s_f.
// ---------------------------------------------------------------------------
__device__ __forceinline__ void wave_gemm(const _Float16* s_w, const _Float16* s_f,
                                          int wave, int lane, v8f acc[2])
{
  int half = lane >> 4;
  int lx   = lane & 15;
  int el   = wave * 16 + lx;          // this lane's B column (edge in tile)

  acc[0] = {};
  acc[1] = {};

#pragma unroll
  for (int ks = 0; ks < 3; ++ks) {
    v16h bf = *(const v16h*)(s_f + el * KPAD + ks * 32 + half * 16);
#pragma unroll
    for (int mt = 0; mt < 2; ++mt) {
      int o = mt * 16 + lx;
      const _Float16* wp = s_w + o * KPAD + ks * 32 + half * 16;
      v8h lo = *(const v8h*)(wp);       // K = kk+16h  .. +7
      v8h hi = *(const v8h*)(wp + 8);   // K = kk+16h+8.. +7
      v16h af;
#pragma unroll
      for (int j = 0; j < 8; ++j) { af[j] = lo[j]; af[j + 8] = hi[j]; }
      acc[mt] = __builtin_amdgcn_wmma_f32_16x16x32_f16(
          false, af, false, bf, (short)0, acc[mt], false, false);
    }
  }
}

// ---------------------------------------------------------------------------
// Pass 1: conv tiles -> per-block (sum, sumsq) partials. Fully ordered
// reductions (shuffle butterfly -> per-wave LDS slots -> ordered sum).
// C/D layout: element r of lane l is channel o = r + 8*(l>>4) (+16*mt),
// edge = base + (l&15).
// ---------------------------------------------------------------------------
__global__ void __launch_bounds__(NTHREADS)
meshconv_pass1(const float* __restrict__ x, const int* __restrict__ gemm,
               const float* __restrict__ wgt, float* __restrict__ part)
{
  __shared__ __align__(32) _Float16 s_w[COUT * KPAD];
  __shared__ __align__(32) _Float16 s_f[TILE_E * KPAD];
  __shared__ int   s_g[TILE_E * 4];
  __shared__ float s_ps[NWAVES * COUT];
  __shared__ float s_pq[NWAVES * COUT];

  int tid = threadIdx.x;
  int b   = blockIdx.y;
  int e0  = blockIdx.x * TILE_E;

  build_tile(x, gemm, wgt, s_w, s_f, s_g, b, e0, tid);

  int wave = tid >> 5, lane = tid & 31;
  v8f acc[2];
  wave_gemm(s_w, s_f, wave, lane, acc);

  int half = lane >> 4, lx = lane & 15;
#pragma unroll
  for (int mt = 0; mt < 2; ++mt) {
#pragma unroll
    for (int r = 0; r < 8; ++r) {
      float s = acc[mt][r];
      float q = s * s;
#pragma unroll
      for (int m = 1; m < 16; m <<= 1) {     // butterfly within 16-lane half
        s += __shfl_xor(s, m, 32);
        q += __shfl_xor(q, m, 32);
      }
      if (lx == 0) {                          // lanes 0 and 16 publish
        int o = mt * 16 + half * 8 + r;
        s_ps[wave * COUT + o] = s;
        s_pq[wave * COUT + o] = q;
      }
    }
  }
  __syncthreads();
  if (tid < COUT) {
    float S = 0.f, Q = 0.f;
#pragma unroll
    for (int w = 0; w < NWAVES; ++w) { S += s_ps[w * COUT + tid]; Q += s_pq[w * COUT + tid]; }
    size_t blk = (size_t)blockIdx.y * gridDim.x + blockIdx.x;
    part[(blk * COUT + tid) * 2 + 0] = S;
    part[(blk * COUT + tid) * 2 + 1] = Q;
  }
}

// ---------------------------------------------------------------------------
// Ordered partial reduction -> stats[0..31] = mean, stats[32..63] = rsqrt(var+eps)
// ---------------------------------------------------------------------------
__global__ void __launch_bounds__(256)
reduce_stats(const float* __restrict__ part, float* __restrict__ stats)
{
  __shared__ float sh[2 * 256];
  int t = threadIdx.x;
  int o = t & 31;
  int chunk = t >> 5;                       // 8 chunks per channel
  int per = (NPART + 7) / 8;
  int p0 = chunk * per;
  int p1 = p0 + per; if (p1 > NPART) p1 = NPART;
  float s = 0.f, q = 0.f;
  for (int p = p0; p < p1; ++p) {
    s += part[((size_t)p * COUT + o) * 2 + 0];
    q += part[((size_t)p * COUT + o) * 2 + 1];
  }
  sh[t] = s; sh[256 + t] = q;
  __syncthreads();
  if (t < COUT) {
    float S = 0.f, Q = 0.f;
#pragma unroll
    for (int c = 0; c < 8; ++c) { S += sh[c * 32 + t]; Q += sh[256 + c * 32 + t]; }
    float n    = (float)Bn * (float)En;     // population stats over (B, E)
    float mean = S / n;
    float var  = Q / n - mean * mean;
    stats[t]        = mean;
    stats[COUT + t] = rsqrtf(var + EPSV);
  }
}

// ---------------------------------------------------------------------------
// Pass 2: recompute conv tiles (L2-resident), fuse BN + ReLU, store.
// ---------------------------------------------------------------------------
__global__ void __launch_bounds__(NTHREADS)
meshconv_pass2(const float* __restrict__ x, const int* __restrict__ gemm,
               const float* __restrict__ wgt, const float* __restrict__ gma,
               const float* __restrict__ bta, const float* __restrict__ stats,
               float* __restrict__ out)
{
  __shared__ __align__(32) _Float16 s_w[COUT * KPAD];
  __shared__ __align__(32) _Float16 s_f[TILE_E * KPAD];
  __shared__ int s_g[TILE_E * 4];

  int tid = threadIdx.x;
  int b   = blockIdx.y;
  int e0  = blockIdx.x * TILE_E;

  build_tile(x, gemm, wgt, s_w, s_f, s_g, b, e0, tid);

  int wave = tid >> 5, lane = tid & 31;
  v8f acc[2];
  wave_gemm(s_w, s_f, wave, lane, acc);       // EXEC all-1s through WMMA

  int half = lane >> 4, lx = lane & 15;
  int e = e0 + wave * 16 + lx;
  if (e < En) {
#pragma unroll
    for (int mt = 0; mt < 2; ++mt) {
#pragma unroll
      for (int r = 0; r < 8; ++r) {
        int o = mt * 16 + half * 8 + r;
        float v = gma[o] * (acc[mt][r] - stats[o]) * stats[COUT + o] + bta[o];
        out[((size_t)b * COUT + o) * En + e] = fmaxf(v, 0.f);
      }
    }
  }
}

// gemm passthrough -> float tail of d_out (tuple output, concatenated flat)
__global__ void __launch_bounds__(256)
copy_gemm(const int* __restrict__ g, float* __restrict__ out, int n)
{
  int i = blockIdx.x * blockDim.x + threadIdx.x;
  if (i < n) out[i] = (float)g[i];
}

// ---------------------------------------------------------------------------
extern "C" void kernel_launch(void* const* d_in, const int* in_sizes, int n_in,
                              void* d_out, int out_size, void* d_ws, size_t ws_size,
                              hipStream_t stream)
{
  const float* x    = (const float*)d_in[0];   // [B, CIN, E]
  const int*   gemm = (const int*)  d_in[1];   // [B, E, 4]
  const float* wgt  = (const float*)d_in[2];   // [COUT, CIN, 5]
  // d_in[3] = bias: cancels exactly under train-mode BatchNorm (y - mean)
  const float* gma  = (const float*)d_in[4];
  const float* bta  = (const float*)d_in[5];
  float* out = (float*)d_out;

  float* part  = (float*)d_ws;                        // NPART*COUT*2 floats (~1.9 MB)
  float* stats = part + (size_t)NPART * COUT * 2;     // 64 floats

  dim3 grid(NBLKX, Bn);
  meshconv_pass1<<<grid, NTHREADS, 0, stream>>>(x, gemm, wgt, part);
  reduce_stats<<<1, 256, 0, stream>>>(part, stats);
  meshconv_pass2<<<grid, NTHREADS, 0, stream>>>(x, gemm, wgt, gma, bta, stats, out);

  int ng = Bn * En * 4;
  copy_gemm<<<(ng + 255) / 256, 256, 0, stream>>>(gemm, out + (size_t)Bn * COUT * En, ng);
}
